// Disc_87703232184815
// MI455X (gfx1250) — matrix operand order repaired and checked by reading, not compiled, not run on hardware
//
#include <hip/hip_runtime.h>

typedef __attribute__((ext_vector_type(16))) _Float16 v16h;
typedef __attribute__((ext_vector_type(8)))  float    v8f;

#define BB   256
#define SS   128
#define DD   256
#define HH   8
#define DHH  32
#define MM   256
#define HIDc 256
#define LL   4

#define NORM_QK   0.4204482076f   /* 32^-0.25            */
#define DIAG_C    0.0883883476f   /* 0.5 * 32^-0.5       */
#define RATIO     0.0625f         /* 256^-0.5            */
#define KEPS      1.0e-4f
#define LNEPS     1.0e-5f

// ---------------- WMMA fragment helpers (CDNA5 16x16x32 f16 layouts) --------

__device__ __forceinline__ v8f wmma16(v16h a, v16h b, v8f c) {
  return __builtin_amdgcn_wmma_f32_16x16x32_f16(false, a, false, b, (short)0, c,
                                                false, false);
}

// A (16x32, f16): lane<16 -> row=lane, K={0..15}; lane>=16 -> row=lane-16,
// K={16..31}.  Source is f32 with strides (rs, ks), scaled + cvt to f16.
__device__ __forceinline__ v16h load_a_f32(const float* __restrict__ p,
                                           int rs, int ks, float scale) {
  const int lane = threadIdx.x & 31;
  const int row  = lane & 15;
  const int kb   = (lane & 16) ? 16 : 0;
  v16h a;
#pragma unroll
  for (int i = 0; i < 8; ++i)
    a[i] = (_Float16)(p[(size_t)row * rs + (size_t)(kb + i) * ks] * scale);
#pragma unroll
  for (int i = 0; i < 8; ++i)
    a[i + 8] = (_Float16)(p[(size_t)row * rs + (size_t)(kb + 8 + i) * ks] * scale);
  return a;
}

// B (32x16, f16) from a row-major [N,K] f16 weight: lane col=lane&15,
// kb=(lane&16)?16:0 -> 16 contiguous halfs from row `col`.
__device__ __forceinline__ v16h load_b_h16(const _Float16* __restrict__ w, int ldk) {
  const int lane = threadIdx.x & 31;
  const int col  = lane & 15;
  const int kb   = (lane & 16) ? 16 : 0;
  const _Float16* p = w + (size_t)col * ldk + kb;
  v16h b;
#pragma unroll
  for (int i = 0; i < 16; ++i) b[i] = p[i];
  return b;
}

// B (32x16) generic from f32 with strides (ks over K, ns over N).
__device__ __forceinline__ v16h load_b_f32(const float* __restrict__ p,
                                           int ks, int ns, float scale) {
  const int lane = threadIdx.x & 31;
  const int col  = lane & 15;
  const int kb   = (lane & 16) ? 16 : 0;
  v16h b;
#pragma unroll
  for (int i = 0; i < 16; ++i)
    b[i] = (_Float16)(p[(size_t)(kb + i) * ks + (size_t)col * ns] * scale);
  return b;
}

// D/C (16x16 f32): VGPR r -> M = r (lane<16) or r+8 (lane>=16), N = lane&15.

// ---------------- generic GEMM:  C = act(A @ W^T + bias) * rowmul -----------
// A: f32 [Mtiles*16 rows, K] with row stride lda.  W: f16 [N,K] row-major.
// Each wave computes a 16x64 strip: one A fragment feeds 4 WMMAs per K-step.
__global__ void gemm_aw(const float* __restrict__ A, int lda,
                        const _Float16* __restrict__ W, int K,
                        const float* __restrict__ bias,
                        const float* __restrict__ rowmul,
                        float* __restrict__ C, int ldc,
                        int Mtiles, int Ngroups, int act) {
  const int wid = (int)((blockIdx.x * blockDim.x + threadIdx.x) >> 5);
  if (wid >= Mtiles * Ngroups) return;
  const int mt = wid / Ngroups, ng = wid % Ngroups;
  const int lane = threadIdx.x & 31;
  v8f acc0 = {}, acc1 = {}, acc2 = {}, acc3 = {};
  const float*    Ab = A + (size_t)(mt * 16) * lda;
  const _Float16* Wb = W + (size_t)(ng * 64) * K;
  for (int kk = 0; kk < K; kk += 32) {
    v16h a = load_a_f32(Ab + kk, lda, 1, 1.0f);
    acc0 = wmma16(a, load_b_h16(Wb + kk, K), acc0);
    acc1 = wmma16(a, load_b_h16(Wb + (size_t)16 * K + kk, K), acc1);
    acc2 = wmma16(a, load_b_h16(Wb + (size_t)32 * K + kk, K), acc2);
    acc3 = wmma16(a, load_b_h16(Wb + (size_t)48 * K + kk, K), acc3);
  }
  v8f accs[4] = {acc0, acc1, acc2, acc3};
#pragma unroll
  for (int j = 0; j < 4; ++j) {
    const int col = ng * 64 + j * 16 + (lane & 15);
    const float bv = bias ? bias[col] : 0.0f;
#pragma unroll
    for (int r = 0; r < 8; ++r) {
      const int row = mt * 16 + ((lane & 16) ? r + 8 : r);
      float x = accs[j][r] + bv;
      if (rowmul) x *= rowmul[row];
      if (act) x = (x >= 0.0f) ? x : 0.2f * x;
      C[(size_t)row * ldc + col] = x;
    }
  }
}

// ---------------- global max of dd_k (WMMA recompute, per-wave partials) ----
// One wave per (b, h, s-tile): A fragment loaded once, swept over 16 proj tiles.
__global__ void kmax_pass(const float* __restrict__ K,
                          const _Float16* __restrict__ proj,
                          float* __restrict__ partial) {
  const int wid = (int)((blockIdx.x * blockDim.x + threadIdx.x) >> 5); // 16384
  const int st = wid & 7;
  const int bh = wid >> 3;
  const int hh = bh & 7, bb = bh >> 3;
  const int lane = threadIdx.x & 31;
  v16h a = load_a_f32(K + ((size_t)(bb * SS + st * 16)) * DD + hh * DHH,
                      DD, 1, NORM_QK);
  float m = -3.4e38f;
  for (int mt = 0; mt < 16; ++mt) {
    v8f acc = {};
    v16h b = load_b_h16(proj + (size_t)(mt * 16) * DHH, DHH);
    acc = wmma16(a, b, acc);
#pragma unroll
    for (int r = 0; r < 8; ++r) m = fmaxf(m, acc[r]);
  }
  for (int o = 16; o; o >>= 1) m = fmaxf(m, __shfl_xor(m, o));
  if (lane == 0) partial[wid] = m;
}

__global__ void reduce_max(const float* __restrict__ p, int n, float* __restrict__ out) {
  __shared__ float sm[1024];
  float m = -3.4e38f;
  for (int i = threadIdx.x; i < n; i += blockDim.x) m = fmaxf(m, p[i]);
  sm[threadIdx.x] = m;
  __syncthreads();
  for (int s = 512; s; s >>= 1) {
    if ((int)threadIdx.x < s) sm[threadIdx.x] = fmaxf(sm[threadIdx.x], sm[threadIdx.x + s]);
    __syncthreads();
  }
  if (threadIdx.x == 0) *out = sm[0];
}

// ---------------- whole Performer attention for one (b,h) in LDS -----------
// LDS: qp[128x256] kp[128x256] ctx[256x32] ksum[256] dinv[128] rmax[128]
//      dq[128] dk[128]  -> 74496 floats = 297984 B (fits 320KB WGP LDS)
__global__ __launch_bounds__(512) void attention_bh(
    const float* __restrict__ q, const float* __restrict__ k,
    const float* __restrict__ v, const _Float16* __restrict__ proj,
    const float* __restrict__ gmax, float* __restrict__ out) {
  extern __shared__ float smem[];
  float* qp   = smem;             // 32768
  float* kp   = qp + 32768;       // 32768
  float* ctxs = kp + 32768;       // 8192
  float* ksum = ctxs + 8192;      // 256
  float* dinv = ksum + 256;       // 128
  float* rmax = dinv + 128;       // 128
  float* dq   = rmax + 128;       // 128
  float* dk   = dq + 128;         // 128

  const int bb = (int)blockIdx.x >> 3;
  const int hh = (int)blockIdx.x & 7;
  const int tid = threadIdx.x;
  const int wave = tid >> 5;
  const int lane = tid & 31;
  const size_t base = ((size_t)bb * SS) * DD + (size_t)hh * DHH;

  // diag(s) = 0.5 * norm^2 * sum_d data^2
  if (tid < 256) {
    const float* src = (tid < 128) ? q : k;
    float* dst = (tid < 128) ? dq : dk;
    const int s = tid & 127;
    const float* p = src + base + (size_t)s * DD;
    float acc = 0.0f;
#pragma unroll
    for (int d = 0; d < DHH; ++d) { const float xv = p[d]; acc += xv * xv; }
    dst[s] = DIAG_C * acc;
  }

  // dd tiles: wave -> (matrix, s-tile); A fragment loaded once, 16 proj tiles
  {
    const bool isk = (wave >= 8);
    const int st = wave & 7;
    const float* src = isk ? k : q;
    float* dst = isk ? kp : qp;
    v16h a = load_a_f32(src + base + (size_t)(st * 16) * DD, DD, 1, NORM_QK);
    for (int mt = 0; mt < 16; ++mt) {
      v8f acc = {};
      v16h b = load_b_h16(proj + (size_t)(mt * 16) * DHH, DHH);
      acc = wmma16(a, b, acc);
#pragma unroll
      for (int r = 0; r < 8; ++r) {
        const int rr = (lane & 16) ? r + 8 : r;
        dst[(st * 16 + rr) * MM + mt * 16 + (lane & 15)] = acc[r];
      }
    }
  }
  __syncthreads();

  // per-row max for q
  if (tid < 128) {
    const float* row = qp + tid * MM;
    float m = row[0];
    for (int i = 1; i < MM; ++i) m = fmaxf(m, row[i]);
    rmax[tid] = m;
  }
  __syncthreads();

  // exponentiate both feature maps
  {
    const float gm = *gmax;
    for (int i = tid; i < SS * MM; i += 512) {
      const int s = i >> 8;
      qp[i] = RATIO * (expf(qp[i] - dq[s] - rmax[s]) + KEPS);
      kp[i] = RATIO * (expf(kp[i] - dk[s] - gm) + KEPS);
    }
  }
  __syncthreads();

  // ksum[m] = sum_s kp[s][m]
  if (tid < 256) {
    float s = 0.0f;
    for (int n = 0; n < SS; ++n) s += kp[n * MM + tid];
    ksum[tid] = s;
  }
  __syncthreads();

  // d_inv[s] = 1 / (qp[s,:] . ksum)
  if (tid < 128) {
    const float* row = qp + tid * MM;
    float s = 0.0f;
    for (int m = 0; m < MM; ++m) s += row[m] * ksum[m];
    dinv[tid] = 1.0f / s;
  }

  // ctx = kp^T @ v: wave -> m-tile; A fragment reused across both d-tiles
  {
    const int mt = wave;
    v8f acc0 = {}, acc1 = {};
    for (int kk = 0; kk < SS; kk += 32) {
      v16h a = load_a_f32(kp + kk * MM + mt * 16, 1, MM, 1.0f);
      acc0 = wmma16(a, load_b_f32(v + base + (size_t)kk * DD, DD, 1, 1.0f), acc0);
      acc1 = wmma16(a, load_b_f32(v + base + (size_t)kk * DD + 16, DD, 1, 1.0f), acc1);
    }
#pragma unroll
    for (int r = 0; r < 8; ++r) {
      const int rr = (lane & 16) ? r + 8 : r;
      ctxs[(mt * 16 + rr) * DHH + (lane & 15)]      = acc0[r];
      ctxs[(mt * 16 + rr) * DHH + 16 + (lane & 15)] = acc1[r];
    }
  }
  __syncthreads();

  // o = (qp @ ctx) * d_inv  -> merged-head layout [b, s, h*32 + d]
  {
    const int st = wave >> 1, dt = wave & 1;
    v8f acc = {};
    for (int kk = 0; kk < MM; kk += 32) {
      v16h a = load_a_f32(qp + (st * 16) * MM + kk, MM, 1, 1.0f);
      v16h b = load_b_f32(ctxs + kk * DHH + dt * 16, DHH, 1, 1.0f);
      acc = wmma16(a, b, acc);
    }
#pragma unroll
    for (int r = 0; r < 8; ++r) {
      const int rr = (lane & 16) ? r + 8 : r;
      const int s = st * 16 + rr;
      out[base + (size_t)s * DD + dt * 16 + (lane & 15)] = acc[r] * dinv[s];
    }
  }
}

// ---------------- small elementwise / reduction kernels ---------------------

__global__ void embed_kernel(const float* __restrict__ x, const float* __restrict__ ew,
                             const float* __restrict__ eb, float* __restrict__ h) {
  const int idx = (int)(blockIdx.x * blockDim.x + threadIdx.x); // B*S*D
  const int d = idx & (DD - 1);
  const int s = (idx >> 8) & (SS - 1);
  const int b = idx >> 15;
  if (b >= BB) return;
  if (s == 0) { h[idx] = 0.0f; return; }
  const float* xp = x + ((size_t)b * (SS - 1) + (s - 1)) * 3;
  h[idx] = xp[0] * ew[d * 3] + xp[1] * ew[d * 3 + 1] + xp[2] * ew[d * 3 + 2] + eb[d];
}

__global__ void keep_kernel(const int* __restrict__ mask, float* __restrict__ keep) {
  const int idx = (int)(blockIdx.x * blockDim.x + threadIdx.x); // B*S
  if (idx >= BB * SS) return;
  const int s = idx & (SS - 1);
  const int b = idx >> 7;
  keep[idx] = (s == 0) ? 1.0f : (mask[b * (SS - 1) + (s - 1)] ? 0.0f : 1.0f);
}

__global__ void cvt_f16(_Float16* __restrict__ dst, const float* __restrict__ src, int n) {
  const int i = (int)(blockIdx.x * blockDim.x + threadIdx.x);
  if (i < n) dst[i] = (_Float16)src[i];
}

__global__ void layernorm_rows(float* __restrict__ xr, const float* __restrict__ g,
                               const float* __restrict__ bta, int rows) {
  const int wid = (int)((blockIdx.x * blockDim.x + threadIdx.x) >> 5);
  if (wid >= rows) return;
  const int lane = threadIdx.x & 31;
  float* row = xr + (size_t)wid * DD;
  float vbuf[8];
  float s = 0.0f;
#pragma unroll
  for (int j = 0; j < 8; ++j) { vbuf[j] = row[lane + 32 * j]; s += vbuf[j]; }
  for (int o = 16; o; o >>= 1) s += __shfl_xor(s, o);
  const float mu = s * (1.0f / DD);
  float qv = 0.0f;
#pragma unroll
  for (int j = 0; j < 8; ++j) { const float dv = vbuf[j] - mu; qv += dv * dv; }
  for (int o = 16; o; o >>= 1) qv += __shfl_xor(qv, o);
  const float inv = rsqrtf(qv * (1.0f / DD) + LNEPS);
#pragma unroll
  for (int j = 0; j < 8; ++j) {
    const int c = lane + 32 * j;
    row[c] = (vbuf[j] - mu) * inv * g[c] + bta[c];
  }
}

__global__ void head_out(const float* __restrict__ c2, const float* __restrict__ ow,
                         const float* __restrict__ ob, float* __restrict__ out) {
  const int wid = (int)((blockIdx.x * blockDim.x + threadIdx.x) >> 5);
  if (wid >= BB) return;
  const int lane = threadIdx.x & 31;
  const float* row = c2 + (size_t)wid * HIDc;
  float s = 0.0f;
  for (int j = lane; j < HIDc; j += 32) s += row[j] * ow[j];
  for (int o = 16; o; o >>= 1) s += __shfl_xor(s, o);
  if (lane == 0) out[wid] = s + ob[0];
}

// ---------------- host orchestration ----------------------------------------

extern "C" void kernel_launch(void* const* d_in, const int* in_sizes, int n_in,
                              void* d_out, int out_size, void* d_ws, size_t ws_size,
                              hipStream_t stream) {
  (void)in_sizes; (void)n_in; (void)out_size; (void)ws_size;
  const float* x     = (const float*)d_in[0];
  const int*   mask  = (const int*)d_in[1];
  const float* emb_w = (const float*)d_in[2];
  const float* emb_b = (const float*)d_in[3];
  const float* Wq    = (const float*)d_in[4];
  const float* bq    = (const float*)d_in[5];
  const float* Wk    = (const float*)d_in[6];
  const float* bk    = (const float*)d_in[7];
  const float* Wv    = (const float*)d_in[8];
  const float* bv    = (const float*)d_in[9];
  const float* Wo    = (const float*)d_in[10];
  const float* bo    = (const float*)d_in[11];
  const float* proj  = (const float*)d_in[12];
  const float* n1w   = (const float*)d_in[13];
  const float* n1b   = (const float*)d_in[14];
  const float* n2w   = (const float*)d_in[15];
  const float* n2b   = (const float*)d_in[16];
  const float* f1w   = (const float*)d_in[17];
  const float* f1b   = (const float*)d_in[18];
  const float* f2w   = (const float*)d_in[19];
  const float* f2b   = (const float*)d_in[20];
  const float* h1w   = (const float*)d_in[21];
  const float* h1b   = (const float*)d_in[22];
  const float* h2w   = (const float*)d_in[23];
  const float* h2b   = (const float*)d_in[24];
  const float* ow    = (const float*)d_in[25];
  const float* ob    = (const float*)d_in[26];

  const size_t SZA = (size_t)BB * SS * DD; // 8388608 floats per activation buffer

  size_t off = 0;
  char* wsb = (char*)d_ws;
  auto take = [&](size_t bytes) -> void* {
    void* p = wsb + off;
    off = (off + bytes + 255) & ~(size_t)255;
    return p;
  };
  float* bufs[4];
  for (int i = 0; i < 4; ++i) bufs[i] = (float*)take(SZA * 4);
  _Float16* wq16   = (_Float16*)take((size_t)LL * DD * DD * 2);
  _Float16* wk16   = (_Float16*)take((size_t)LL * DD * DD * 2);
  _Float16* wv16   = (_Float16*)take((size_t)LL * DD * DD * 2);
  _Float16* wo16   = (_Float16*)take((size_t)LL * DD * DD * 2);
  _Float16* f1w16  = (_Float16*)take((size_t)LL * HIDc * DD * 2);
  _Float16* f2w16  = (_Float16*)take((size_t)LL * DD * HIDc * 2);
  _Float16* proj16 = (_Float16*)take((size_t)LL * MM * DHH * 2);
  _Float16* h1w16  = (_Float16*)take((size_t)2 * HIDc * DD * 2);
  _Float16* h2w16  = (_Float16*)take((size_t)HIDc * 2 * HIDc * 2);
  float* keepv   = (float*)take((size_t)BB * SS * 4);
  float* partial = (float*)take((size_t)16384 * 4);
  float* gmax    = (float*)take(256);
  float* c1      = (float*)take((size_t)BB * 2 * HIDc * 4);
  float* c2      = (float*)take((size_t)BB * HIDc * 4);

  auto cvt = [&](_Float16* dst, const float* src, int n) {
    cvt_f16<<<(n + 255) / 256, 256, 0, stream>>>(dst, src, n);
  };
  cvt(wq16, Wq, LL * DD * DD);
  cvt(wk16, Wk, LL * DD * DD);
  cvt(wv16, Wv, LL * DD * DD);
  cvt(wo16, Wo, LL * DD * DD);
  cvt(f1w16, f1w, LL * HIDc * DD);
  cvt(f2w16, f2w, LL * DD * HIDc);
  cvt(proj16, proj, LL * MM * DHH);
  cvt(h1w16, h1w, 2 * HIDc * DD);
  cvt(h2w16, h2w, HIDc * 2 * HIDc);

  embed_kernel<<<(int)(SZA / 256), 256, 0, stream>>>(x, emb_w, emb_b, bufs[0]);
  keep_kernel<<<(BB * SS) / 256, 256, 0, stream>>>(mask, keepv);

  auto gemm = [&](const float* A, int lda, const _Float16* W, int K,
                  const float* bias, const float* rowmul, float* C, int ldc,
                  int Mt, int Ng, int act) {
    const int waves = Mt * Ng;
    gemm_aw<<<(waves * 32 + 255) / 256, 256, 0, stream>>>(A, lda, W, K, bias,
                                                          rowmul, C, ldc, Mt, Ng, act);
  };

  const size_t ATT_SMEM = (size_t)(32768 + 32768 + 8192 + 256 + 128 * 4) * 4;
  const int rowsAll = BB * SS;        // 32768
  const int MtAll = rowsAll / 16;     // 2048

  int hi = 0;
  for (int l = 0; l < LL; ++l) {
    float* hcur = bufs[hi];
    float* qb   = bufs[(hi + 1) & 3];
    float* kb2  = bufs[(hi + 2) & 3];
    float* vb   = bufs[(hi + 3) & 3];
    const _Float16* wq_l = wq16 + (size_t)l * DD * DD;
    const _Float16* wk_l = wk16 + (size_t)l * DD * DD;
    const _Float16* wv_l = wv16 + (size_t)l * DD * DD;
    const _Float16* wo_l = wo16 + (size_t)l * DD * DD;
    const _Float16* pj_l = proj16 + (size_t)l * MM * DHH;

    gemm(hcur, DD, wq_l, DD, bq + l * DD, nullptr, qb, DD, MtAll, DD / 64, 0);
    gemm(hcur, DD, wk_l, DD, bk + l * DD, nullptr, kb2, DD, MtAll, DD / 64, 0);
    gemm(hcur, DD, wv_l, DD, bv + l * DD, keepv, vb, DD, MtAll, DD / 64, 0);

    kmax_pass<<<(16384 * 32) / 256, 256, 0, stream>>>(kb2, pj_l, partial);
    reduce_max<<<1, 1024, 0, stream>>>(partial, 16384, gmax);

    attention_bh<<<BB * HH, 512, ATT_SMEM, stream>>>(qb, kb2, vb, pj_l, gmax, hcur);

    gemm(hcur, DD, wo_l, DD, bo + l * DD, nullptr, qb, DD, MtAll, DD / 64, 0);
    layernorm_rows<<<(rowsAll * 32) / 256, 256, 0, stream>>>(qb, n1w + l * DD,
                                                             n1b + l * DD, rowsAll);
    gemm(qb, DD, f1w16 + (size_t)l * HIDc * DD, DD, f1b + l * HIDc, nullptr,
         kb2, HIDc, MtAll, HIDc / 64, 1);
    gemm(kb2, HIDc, f2w16 + (size_t)l * DD * HIDc, HIDc, f2b + l * DD, nullptr,
         vb, DD, MtAll, DD / 64, 0);
    layernorm_rows<<<(rowsAll * 32) / 256, 256, 0, stream>>>(vb, n2w + l * DD,
                                                             n2b + l * DD, rowsAll);
    hi = (hi + 3) & 3;
  }

  // classification head: CLS rows are h[b, 0, :] -> A with row stride S*D
  float* hfin = bufs[hi];
  gemm(hfin, SS * DD, h1w16, DD, h1b, nullptr, c1, 2 * HIDc, BB / 16,
       (2 * HIDc) / 64, 1);
  gemm(c1, 2 * HIDc, h2w16, 2 * HIDc, h2b, nullptr, c2, HIDc, BB / 16,
       HIDc / 64, 1);
  head_out<<<(BB * 32) / 256, 256, 0, stream>>>(c2, ow, ob, (float*)d_out);
}
